// TransformerBlock_23605140259128
// MI455X (gfx1250) — compile-verified, matrix operand order checked by reading, not run on hardware
//
#include <hip/hip_runtime.h>
#include <hip/hip_bf16.h>

#define Bb    2
#define DIMC  192
#define HEADS 4
#define DHd   48
#define KP2   9
#define HIDC  510
#define Hh    128
#define Ww    128
#define HW    16384

typedef __attribute__((ext_vector_type(16))) __bf16 v16bf;
typedef __attribute__((ext_vector_type(8)))  float  v8f;

__device__ __forceinline__ unsigned short f2bfu(float f) {
  union { float f; unsigned u; } un; un.f = f;
  unsigned u = un.u;
  u += 0x7FFFu + ((u >> 16) & 1u);          // round-to-nearest-even
  return (unsigned short)(u >> 16);
}
__device__ __forceinline__ __bf16 f2bf(float f) {
  unsigned short s = f2bfu(f);
  union { unsigned short s; __bf16 b; } cv; cv.s = s;
  return cv.b;
}

// ---------------------------------------------------------------------------
// Bias-free LayerNorm over channel dim: y = x * rsqrt(var + 1e-5) * w
// ---------------------------------------------------------------------------
__global__ void ln_kernel(const float* __restrict__ x,
                          const float* __restrict__ w,
                          float* __restrict__ out) {
  int p = blockIdx.x * blockDim.x + threadIdx.x;   // pixel
  int b = blockIdx.y;
  if (p >= HW) return;
  const float* xb = x + (unsigned)b * DIMC * HW;
  float s = 0.f, ss = 0.f;
  for (int c = 0; c < DIMC; ++c) {
    float v = xb[(unsigned)c * HW + p];
    s += v; ss += v * v;
  }
  float mu  = s * (1.0f / DIMC);
  float var = ss * (1.0f / DIMC) - mu * mu;
  float r   = rsqrtf(var + 1e-5f);
  float* ob = out + (unsigned)b * DIMC * HW;
  for (int c = 0; c < DIMC; ++c)
    ob[(unsigned)c * HW + p] = xb[(unsigned)c * HW + p] * r * w[c];
}

// ---------------------------------------------------------------------------
// Out[m,n] = sum_k Wm[m,k] * X[k,n]  (+ Res[m,n]) per batch (blockIdx.z)
// bf16 WMMA, f32 accumulate. BM=64, BN=64, BK=32; 8 waves/block, each wave
// computes one 16(M) x 32(N) strip = 2 x v_wmma_f32_16x16x32_bf16 per K-chunk.
// LDS tiles stored pre-swizzled in WMMA fragment layout (2x ds_load_b128 per
// fragment), double-buffered so chunk c+1 global loads overlap chunk c WMMAs.
// Fragment element mapping (ISA 7.12.2, 16-bit A/B, 16x32):
//   lane = (row&15) | (((k>>3)&1)<<4)
//   e    = (((k>>4)&1)*4 + ((k>>1)&3))*2 + (k&1)
// ---------------------------------------------------------------------------
#define BM 64
#define BN 64
#define BK 32

__global__ __launch_bounds__(256)
void gemm_bf16_kernel(const float* __restrict__ Wm,   // [M,K] (shared by batch)
                      const float* __restrict__ X,    // [B,K,N]
                      float* __restrict__ Out,        // [B,M,N]
                      const float* __restrict__ Res,  // [B,M,N] or null
                      int M, int K, int N) {
  // double-buffered, [4 subtiles][32 lanes][16 halfs] each
  __shared__ __bf16 Af[2][4 * 32 * 16];
  __shared__ __bf16 Bf[2][4 * 32 * 16];

  int b = blockIdx.z;
  const float* Xb = X + (unsigned)b * (unsigned)(K * N);
  float*       Ob = Out + (unsigned)b * (unsigned)(M * N);
  const float* Rb = Res ? Res + (unsigned)b * (unsigned)(M * N) : nullptr;

  int m0 = blockIdx.y * BM;
  int n0 = blockIdx.x * BN;
  int tid  = threadIdx.x;
  int lane = tid & 31;
  int wave = tid >> 5;
  int sr = wave & 3;              // A subtile (16 rows of M)
  int sc = (wave >> 2) * 2;       // first of two B col-subtiles

  v8f acc0 = {}; v8f acc1 = {};

  // staging: A = 4 float2 (pairs along K), B = 2 float4 (quads along N)
  float2 wa[4]; bool oa[4];
  float4 xa[2]; bool ox[2];

  auto stageA_load = [&](int k0) {
#pragma unroll
    for (int it = 0; it < 4; ++it) {
      int i = tid + it * 256;
      int mm = i >> 4, kk = (i & 15) * 2;
      int gm = m0 + mm, gk = k0 + kk;
      int gmc = min(gm, M - 1), gkc = min(gk, K - 2);  // K even at all call sites
      wa[it] = *(const float2*)&Wm[(unsigned)gmc * K + gkc];
      oa[it] = (gm >= M) | (gk >= K);
    }
  };
  auto stageB_load = [&](int k0) {
#pragma unroll
    for (int it = 0; it < 2; ++it) {
      int i = tid + it * 256;
      int kk = i >> 4, nn = (i & 15) * 4;
      int gk = k0 + kk, gkc = min(gk, K - 1);
      xa[it] = *(const float4*)&Xb[(unsigned)gkc * N + n0 + nn];
      ox[it] = (gk >= K);
    }
  };
  auto stageA_store = [&](__bf16* A) {
#pragma unroll
    for (int it = 0; it < 4; ++it) {
      int i = tid + it * 256;
      int mm = i >> 4, kk = (i & 15) * 2;
      unsigned lo = f2bfu(oa[it] ? 0.f : wa[it].x);
      unsigned hi = f2bfu(oa[it] ? 0.f : wa[it].y);
      int lw = (mm & 15) | (((kk >> 3) & 1) << 4);
      int e  = ((((kk >> 4) & 1) * 4 + ((kk >> 1) & 3)) << 1);  // even
      *(unsigned*)&A[((mm >> 4) << 9) + (lw << 4) + e] = lo | (hi << 16);
    }
  };
  auto stageB_store = [&](__bf16* Bd) {
#pragma unroll
    for (int it = 0; it < 2; ++it) {
      int i = tid + it * 256;
      int kk = i >> 4, nn = (i & 15) * 4;
      int e   = (((kk >> 4) & 1) * 4 + ((kk >> 1) & 3)) * 2 + (kk & 1);
      int lhi = ((kk >> 3) & 1) << 4;
      float vv[4] = { xa[it].x, xa[it].y, xa[it].z, xa[it].w };
#pragma unroll
      for (int j = 0; j < 4; ++j) {
        int n = nn + j;
        float f = ox[it] ? 0.f : vv[j];
        Bd[((n >> 4) << 9) + (((n & 15) | lhi) << 4) + e] = f2bf(f);
      }
    }
  };

  int NC = (K + BK - 1) / BK;

  // prologue: stage chunk 0 into buffer 0
  stageA_load(0); stageB_load(0);
  stageA_store(Af[0]); stageB_store(Bf[0]);
  __syncthreads();

  for (int c = 0; c < NC; ++c) {
    int cur = c & 1, nxt = cur ^ 1;
    bool more = (c + 1) < NC;

    if (more) { stageA_load((c + 1) * BK); stageB_load((c + 1) * BK); }

    v16bf a   = *(const v16bf*)&Af[cur][(sr << 9) + (lane << 4)];
    v16bf bb0 = *(const v16bf*)&Bf[cur][(sc << 9) + (lane << 4)];
    v16bf bb1 = *(const v16bf*)&Bf[cur][((sc + 1) << 9) + (lane << 4)];

    if (more) { stageA_store(Af[nxt]); stageB_store(Bf[nxt]); }

    acc0 = __builtin_amdgcn_wmma_f32_16x16x32_bf16(false, a, false, bb0,
                                                   (short)0, acc0, false, false);
    acc1 = __builtin_amdgcn_wmma_f32_16x16x32_bf16(false, a, false, bb1,
                                                   (short)0, acc1, false, false);
    __syncthreads();
  }

  // C/D layout: VGPR r -> M = r + (lane>=16 ? 8 : 0), N = lane&15
  int nCol = lane & 15;
  int mOff = (lane >> 4) ? 8 : 0;
#pragma unroll
  for (int r = 0; r < 8; ++r) {
    int m = m0 + sr * 16 + mOff + r;
    if (m < M) {
      unsigned i0 = (unsigned)m * N + (unsigned)(n0 + sc * 16 + nCol);
      unsigned i1 = i0 + 16;
      float v0 = acc0[r]; float v1 = acc1[r];
      if (Rb) { v0 += Rb[i0]; v1 += Rb[i1]; }
      Ob[i0] = v0; Ob[i1] = v1;
    }
  }
}

// ---------------------------------------------------------------------------
// Deformable attention: one thread per (b, head, pixel)
// qkv: [B, 576, HW]  off: [B, 72, HW]  rpb: [4,9]  out: [B, 192, HW]
// ---------------------------------------------------------------------------
__global__ void deform_attn_kernel(const float* __restrict__ qkv,
                                   const float* __restrict__ off,
                                   const float* __restrict__ rpb,
                                   float* __restrict__ out) {
  int p = blockIdx.x * blockDim.x + threadIdx.x;
  if (p >= HW) return;
  int h = blockIdx.y;
  int b = blockIdx.z;

  const float* qkvb = qkv + (unsigned)b * 3 * DIMC * HW;
  const float* q  = qkvb + (unsigned)(h * DHd) * HW;
  const float* kf = qkvb + (unsigned)(DIMC + h * DHd) * HW;
  const float* vf = qkvb + (unsigned)(2 * DIMC + h * DHd) * HW;
  const float* ob = off + (unsigned)b * (HEADS * KP2 * 2) * HW;

  int py = p >> 7, px = p & 127;

  int   i0[KP2], i1[KP2], i2[KP2], i3[KP2];
  float w00[KP2], w01[KP2], w10[KP2], w11[KP2];
  float logits[KP2];

#pragma unroll
  for (int kp = 0; kp < KP2; ++kp) {
    int ky = kp / 3 - 1, kx = kp % 3 - 1;
    float oy = ob[(unsigned)((h * KP2 + kp) * 2 + 0) * HW + p];
    float ox = ob[(unsigned)((h * KP2 + kp) * 2 + 1) * HW + p];
    float cy = fminf(fmaxf((float)(py + ky) + oy, 0.f), (float)(Hh - 1));
    float cx = fminf(fmaxf((float)(px + kx) + ox, 0.f), (float)(Ww - 1));
    float y0f = floorf(cy), x0f = floorf(cx);
    float wy = cy - y0f, wx = cx - x0f;
    int y0 = (int)y0f, x0 = (int)x0f;
    int y1 = min(y0 + 1, Hh - 1), x1 = min(x0 + 1, Ww - 1);
    i0[kp] = y0 * Ww + x0; i1[kp] = y0 * Ww + x1;
    i2[kp] = y1 * Ww + x0; i3[kp] = y1 * Ww + x1;
    w00[kp] = (1.f - wy) * (1.f - wx); w01[kp] = (1.f - wy) * wx;
    w10[kp] = wy * (1.f - wx);         w11[kp] = wy * wx;
    logits[kp] = 0.f;
  }

  // pass 1: logits[kp] = q . bilerp(k)   (kp fully unrolled -> register arrays)
  for (int d = 0; d < DHd; ++d) {
    float qd = q[(unsigned)d * HW + p];
    const float* kd = kf + (unsigned)d * HW;
#pragma unroll
    for (int kp = 0; kp < KP2; ++kp) {
      float ks = kd[i0[kp]] * w00[kp] + kd[i1[kp]] * w01[kp]
               + kd[i2[kp]] * w10[kp] + kd[i3[kp]] * w11[kp];
      logits[kp] += qd * ks;
    }
  }
  float scale = rsqrtf((float)DHd);
#pragma unroll
  for (int kp = 0; kp < KP2; ++kp)
    logits[kp] = logits[kp] * scale + rpb[h * KP2 + kp];

  // softmax over 9
  float mx = logits[0];
#pragma unroll
  for (int kp = 1; kp < KP2; ++kp) mx = fmaxf(mx, logits[kp]);
  float a[KP2]; float se = 0.f;
#pragma unroll
  for (int kp = 0; kp < KP2; ++kp) { a[kp] = __expf(logits[kp] - mx); se += a[kp]; }
  float inv = 1.f / se;
#pragma unroll
  for (int kp = 0; kp < KP2; ++kp) a[kp] *= inv;

  // pass 2: out[d] = sum_kp a[kp] * bilerp(v)
  float* outb = out + (unsigned)b * DIMC * HW + (unsigned)(h * DHd) * HW;
  for (int d = 0; d < DHd; ++d) {
    const float* vd = vf + (unsigned)d * HW;
    float o = 0.f;
#pragma unroll
    for (int kp = 0; kp < KP2; ++kp) {
      float vs = vd[i0[kp]] * w00[kp] + vd[i1[kp]] * w01[kp]
               + vd[i2[kp]] * w10[kp] + vd[i3[kp]] * w11[kp];
      o += a[kp] * vs;
    }
    outb[(unsigned)d * HW + p] = o;
  }
}

// ---------------------------------------------------------------------------
// Depthwise 3x3 (SAME, zero pad) on 1020 channels + gelu(t1)*t2 gate
// t: [B, 1020, HW]  dw: [1020, 9]  g: [B, 510, HW]
// ---------------------------------------------------------------------------
__global__ void dwconv_gelu_kernel(const float* __restrict__ t,
                                   const float* __restrict__ dw,
                                   float* __restrict__ g) {
  int p = blockIdx.x * blockDim.x + threadIdx.x;
  if (p >= HW) return;
  int c = blockIdx.y;   // 0..509
  int b = blockIdx.z;
  int py = p >> 7, px = p & 127;

  const float* t1 = t + (unsigned)b * 2 * HIDC * HW + (unsigned)c * HW;
  const float* t2 = t1 + (unsigned)HIDC * HW;
  const float* w1 = dw + (unsigned)c * 9;
  const float* w2 = dw + (unsigned)(c + HIDC) * 9;

  float d1 = 0.f, d2 = 0.f;
#pragma unroll
  for (int ky = -1; ky <= 1; ++ky) {
#pragma unroll
    for (int kx = -1; kx <= 1; ++kx) {
      int yy = py + ky, xx = px + kx;
      if (yy < 0 || yy >= Hh || xx < 0 || xx >= Ww) continue;
      int idx = yy * Ww + xx;
      int wi = (ky + 1) * 3 + (kx + 1);
      d1 += w1[wi] * t1[idx];
      d2 += w2[wi] * t2[idx];
    }
  }
  float ge = 0.5f * d1 * (1.f + erff(d1 * 0.70710678118654752f));
  g[(unsigned)b * HIDC * HW + (unsigned)c * HW + p] = ge * d2;
}

// ---------------------------------------------------------------------------
extern "C" void kernel_launch(void* const* d_in, const int* in_sizes, int n_in,
                              void* d_out, int out_size, void* d_ws, size_t ws_size,
                              hipStream_t stream) {
  const float* x      = (const float*)d_in[0];
  const float* ln1_w  = (const float*)d_in[1];
  const float* ln2_w  = (const float*)d_in[2];
  const float* qkv_w  = (const float*)d_in[3];
  const float* off_w  = (const float*)d_in[4];
  const float* rpb    = (const float*)d_in[5];
  const float* proj_w = (const float*)d_in[6];
  const float* pi_w   = (const float*)d_in[7];
  const float* dw_w   = (const float*)d_in[8];
  const float* po_w   = (const float*)d_in[9];
  float* out = (float*)d_out;

  // Workspace layout (floats), region A reused by liveness:
  //   xn  : [B,192,HW]                              (LN output, reused for LN2)
  //   A+0 : qkv [B,576,HW]  -> later t [B,1020,HW]
  //   A+576*B*HW : off [B,72,HW]
  //   A+648*B*HW : attn [B,192,HW]
  //   A+1020*B*HW : g [B,510,HW]
  float* ws   = (float*)d_ws;
  float* xn   = ws;
  float* regA = ws + (size_t)Bb * DIMC * HW;
  float* qkv  = regA;
  float* offo = regA + (size_t)Bb * 3 * DIMC * HW;
  float* attn = offo + (size_t)Bb * (HEADS * KP2 * 2) * HW;
  float* tbuf = regA;
  float* gbuf = regA + (size_t)Bb * 2 * HIDC * HW;

  dim3 blk(256);

  // 1) LN1: x -> xn
  ln_kernel<<<dim3(HW / 256, Bb), blk, 0, stream>>>(x, ln1_w, xn);

  // 2) QKV gemm: [576,192] x [192,HW] -> qkv
  gemm_bf16_kernel<<<dim3(HW / BN, (3 * DIMC) / BM, Bb), blk, 0, stream>>>(
      qkv_w, xn, qkv, nullptr, 3 * DIMC, DIMC, HW);

  // 3) offsets gemm: [72,192] x [192,HW] -> off
  gemm_bf16_kernel<<<dim3(HW / BN, (72 + BM - 1) / BM, Bb), blk, 0, stream>>>(
      off_w, xn, offo, nullptr, HEADS * KP2 * 2, DIMC, HW);

  // 4) deformable attention -> attn
  deform_attn_kernel<<<dim3(HW / 256, HEADS, Bb), blk, 0, stream>>>(
      qkv, offo, rpb, attn);

  // 5) proj gemm + residual x -> out  (out = x1)
  gemm_bf16_kernel<<<dim3(HW / BN, DIMC / BM, Bb), blk, 0, stream>>>(
      proj_w, attn, out, x, DIMC, DIMC, HW);

  // 6) LN2: out -> xn
  ln_kernel<<<dim3(HW / 256, Bb), blk, 0, stream>>>(out, ln2_w, xn);

  // 7) FFN in gemm: [1020,192] x [192,HW] -> t  (overwrites dead qkv/off/attn)
  gemm_bf16_kernel<<<dim3(HW / BN, (2 * HIDC + BM - 1) / BM, Bb), blk, 0, stream>>>(
      pi_w, xn, tbuf, nullptr, 2 * HIDC, DIMC, HW);

  // 8) depthwise 3x3 + gelu gate -> g
  dwconv_gelu_kernel<<<dim3(HW / 256, HIDC, Bb), blk, 0, stream>>>(
      tbuf, dw_w, gbuf);

  // 9) FFN out gemm: [192,510] x [510,HW] + out -> out (in-place residual)
  gemm_bf16_kernel<<<dim3(HW / BN, DIMC / BM, Bb), blk, 0, stream>>>(
      po_w, gbuf, out, out, DIMC, HIDC, HW);
}
